// MultiHeadAttention_67697274519607
// MI455X (gfx1250) — matrix operand order repaired and checked by reading, not compiled, not run on hardware
//
#include <hip/hip_runtime.h>
#include <hip/hip_bf16.h>

#define B_ 2
#define S_ 2048
#define D_ 1024
#define H_ 16
#define DH_ 64

typedef __attribute__((ext_vector_type(16))) __bf16 v16bf;
typedef __attribute__((ext_vector_type(8)))  float  v8f;

union FragAB { uint4 q[2]; v16bf v; };
union U4U16  { uint4 q; unsigned short h[8]; };

__device__ __forceinline__ unsigned short f2bf(float f) {
  unsigned int u = __builtin_bit_cast(unsigned int, f);
  u += 0x7FFFu + ((u >> 16) & 1u);          // round-to-nearest-even
  return (unsigned short)(u >> 16);
}

__device__ __forceinline__ v8f wmma_bf16(v16bf a, v16bf b, v8f c) {
  // (neg_a, A, neg_b, B, c_mod, C, reuse_a, reuse_b)
  return __builtin_amdgcn_wmma_f32_16x16x32_bf16(false, a, false, b, (short)0, c,
                                                 false, false);
}

// CDNA5 async global->LDS copy, 16 bytes per lane. Tracked with ASYNCcnt.
// VDST = LDS byte address (low 32 bits of flat address), VADDR = global addr.
__device__ __forceinline__ void async_copy16(unsigned short* ldst,
                                             const unsigned short* gsrc) {
  unsigned int lds = (unsigned int)(unsigned long long)ldst;
  asm volatile("global_load_async_to_lds_b128 %0, %1, off"
               :: "v"(lds), "v"(gsrc) : "memory");
}
__device__ __forceinline__ void wait_async0() {
  asm volatile("s_wait_asynccnt 0x0" ::: "memory");
}

// ---------------------------------------------------------------- convert ---
__global__ void k_convert_bf16(const float* __restrict__ src,
                               unsigned short* __restrict__ dst, int n8) {
  int i = blockIdx.x * blockDim.x + threadIdx.x;
  if (i >= n8) return;
  const float4* s = (const float4*)src + (size_t)i * 2;
  float4 f0 = s[0], f1 = s[1];
  U4U16 o;
  o.h[0] = f2bf(f0.x); o.h[1] = f2bf(f0.y); o.h[2] = f2bf(f0.z); o.h[3] = f2bf(f0.w);
  o.h[4] = f2bf(f1.x); o.h[5] = f2bf(f1.y); o.h[6] = f2bf(f1.z); o.h[7] = f2bf(f1.w);
  ((uint4*)dst)[i] = o.q;
}

// ------------------------------------------ convert + transpose (weights) ---
// dst[n][k] = bf16(src[k][n]), 1024x1024, 32x32 tiles via LDS
__global__ __launch_bounds__(256)
void k_convT_bf16(const float* __restrict__ src,
                  unsigned short* __restrict__ dst) {
  __shared__ unsigned short sT[32][33];
  const int t  = threadIdx.x;
  const int k0 = blockIdx.x * 32, n0 = blockIdx.y * 32;
  const int r  = t >> 3, c4 = (t & 7) * 4;
  float4 f = *(const float4*)(src + (size_t)(k0 + r) * 1024 + n0 + c4);
  sT[c4 + 0][r] = f2bf(f.x);
  sT[c4 + 1][r] = f2bf(f.y);
  sT[c4 + 2][r] = f2bf(f.z);
  sT[c4 + 3][r] = f2bf(f.w);
  __syncthreads();
  unsigned int lo = (unsigned int)sT[r][c4 + 0] | ((unsigned int)sT[r][c4 + 1] << 16);
  unsigned int hi = (unsigned int)sT[r][c4 + 2] | ((unsigned int)sT[r][c4 + 3] << 16);
  uint2 o; o.x = lo; o.y = hi;
  *(uint2*)(dst + (size_t)(n0 + r) * 1024 + k0 + c4) = o;
}

// ------------------------------------------------------------------- GEMM ---
// C[M=4096][N=1024] = A(bf16,[M][1024]) @ W, with Wt = W^T in [N][K] layout.
// mode 0: bf16 out in [B,H,S,DH]   (Q, K projections)
// mode 2: bf16 out in [B,H,DH,S]   (V projection, pre-transposed for attn)
// mode 1: f32 row-major + bias     (output projection)
// Block tile 128x128, BK=64, double-buffered LDS, async global->LDS staging.
__global__ __launch_bounds__(256)
void k_gemm(const unsigned short* __restrict__ A,
            const unsigned short* __restrict__ Wt,
            const float* __restrict__ bias,
            unsigned short* __restrict__ outH,
            float* __restrict__ outF,
            int mode) {
  __shared__ unsigned short sA[2][128 * 64];   // 2 x 16 KB
  __shared__ unsigned short sW[2][128 * 64];   // 2 x 16 KB

  const int tid  = threadIdx.x;
  const int m0   = blockIdx.x * 128;
  const int n0   = blockIdx.y * 128;
  const int w    = tid >> 5, lane = tid & 31;
  const int wm   = w >> 1,  wn   = w & 1;
  const int half = lane >> 4, c  = lane & 15;

  v8f acc[2][4] = {};

  // stage K-tile 0 (async DMA into LDS)
#pragma unroll
  for (int r = 0; r < 4; ++r) {
    int cid = tid + r * 256;
    int row = cid >> 3, cc = cid & 7;
    async_copy16(&sA[0][row * 64 + cc * 8],
                 A  + (size_t)(m0 + row) * 1024 + cc * 8);
    async_copy16(&sW[0][row * 64 + cc * 8],
                 Wt + (size_t)(n0 + row) * 1024 + cc * 8);
  }
  wait_async0();
  __syncthreads();

  for (int kt = 0; kt < 16; ++kt) {
    const int cur = kt & 1;
    const bool pf = (kt < 15);
    if (pf) {                       // issue next-tile async copies early
#pragma unroll
      for (int r = 0; r < 4; ++r) {
        int cid = tid + r * 256;
        int row = cid >> 3, cc = cid & 7;
        async_copy16(&sA[cur ^ 1][row * 64 + cc * 8],
                     A  + (size_t)(m0 + row) * 1024 + (kt + 1) * 64 + cc * 8);
        async_copy16(&sW[cur ^ 1][row * 64 + cc * 8],
                     Wt + (size_t)(n0 + row) * 1024 + (kt + 1) * 64 + cc * 8);
      }
    }
#pragma unroll
    for (int ks = 0; ks < 2; ++ks) {
      FragAB a[2], b[4];
#pragma unroll
      for (int mi = 0; mi < 2; ++mi) {
        int row = wm * 32 + mi * 16 + c;
        a[mi].q[0] = *(const uint4*)(&sA[cur][row * 64 + ks * 32 + half * 8]);
        a[mi].q[1] = *(const uint4*)(&sA[cur][row * 64 + ks * 32 + 16 + half * 8]);
      }
#pragma unroll
      for (int ni = 0; ni < 4; ++ni) {
        int n = wn * 64 + ni * 16 + c;
        b[ni].q[0] = *(const uint4*)(&sW[cur][n * 64 + ks * 32 + half * 8]);
        b[ni].q[1] = *(const uint4*)(&sW[cur][n * 64 + ks * 32 + 16 + half * 8]);
      }
#pragma unroll
      for (int mi = 0; mi < 2; ++mi)
#pragma unroll
        for (int ni = 0; ni < 4; ++ni)
          acc[mi][ni] = wmma_bf16(a[mi].v, b[ni].v, acc[mi][ni]);
    }
    if (pf) wait_async0();          // next tile resident before publishing
    __syncthreads();
  }

#pragma unroll
  for (int mi = 0; mi < 2; ++mi)
#pragma unroll
    for (int ni = 0; ni < 4; ++ni)
#pragma unroll
      for (int i = 0; i < 8; ++i) {
        int m = m0 + wm * 32 + mi * 16 + i + 8 * half;
        int n = n0 + wn * 64 + ni * 16 + c;
        float v = acc[mi][ni][i];
        if (mode == 1) {
          outF[(size_t)m * 1024 + n] = v + bias[n];
        } else {
          int bb = m >> 11, s = m & (S_ - 1);
          int h = n >> 6, dh = n & (DH_ - 1);
          if (mode == 0)
            outH[((((size_t)bb * H_ + h) * S_ + s) << 6) + dh] = f2bf(v);
          else  // mode 2: V pre-transposed [B,H,DH,S]
            outH[(((size_t)bb * H_ + h) * DH_ + dh) * S_ + s] = f2bf(v);
        }
      }
}

// -------------------------------------------------------- flash attention ---
// One block = (b,h) x 64 q-rows. 4 waves, each wave owns 16 q-rows.
// V arrives pre-transposed [B,H,DH,S]; KV tiles staged via async->LDS.
__global__ __launch_bounds__(128)
void k_attn(const unsigned short* __restrict__ Q,
            const unsigned short* __restrict__ K,
            const unsigned short* __restrict__ Vt,
            unsigned short* __restrict__ ctx) {
  __shared__ unsigned short sK[2][32 * 64];     // [kv][dh]  2 x 4 KB
  __shared__ unsigned short sVt[2][64 * 32];    // [dh][kv]  2 x 4 KB
  __shared__ unsigned short sP[4][16 * 32];     // per-wave P     4 KB

  const int tid  = threadIdx.x;
  const int w    = tid >> 5, lane = tid & 31;
  const int half = lane >> 4, c  = lane & 15;
  const int qb   = blockIdx.x;
  const int bh   = blockIdx.y;

  const unsigned short* Qh  = Q  + (size_t)bh * S_ * DH_;
  const unsigned short* Kh  = K  + (size_t)bh * S_ * DH_;
  const unsigned short* Vth = Vt + (size_t)bh * DH_ * S_;

  const int q0 = qb * 64 + w * 16;

  // Q fragments for both DH k-steps, loaded once
  FragAB aQ[2];
  {
    const unsigned short* qp = Qh + (size_t)(q0 + c) * DH_;
#pragma unroll
    for (int ks = 0; ks < 2; ++ks) {
      aQ[ks].q[0] = *(const uint4*)(qp + ks * 32 + half * 8);
      aQ[ks].q[1] = *(const uint4*)(qp + ks * 32 + 16 + half * 8);
    }
  }

  v8f O[4] = {};
  v8f mrow, lrow;
#pragma unroll
  for (int i = 0; i < 8; ++i) { mrow[i] = -3.0e38f; lrow[i] = 0.0f; }

  const int ntiles = 2 * (qb + 1);

  // stage KV tile 0 (async DMA)
#pragma unroll
  for (int r = 0; r < 2; ++r) {
    int cid = tid + r * 128;
    int krow = cid >> 3, kc = cid & 7;
    async_copy16(&sK[0][krow * 64 + kc * 8], Kh + (size_t)krow * DH_ + kc * 8);
    int vrow = cid >> 2, vc = cid & 3;
    async_copy16(&sVt[0][vrow * 32 + vc * 8], Vth + (size_t)vrow * S_ + vc * 8);
  }
  wait_async0();
  __syncthreads();

  for (int kt = 0; kt < ntiles; ++kt) {
    const int cur = kt & 1;
    const int kvb = kt * 32;
    const bool pf = (kt + 1 < ntiles);
    if (pf) {                        // issue next-tile async copies early
      const int nb = kvb + 32;
#pragma unroll
      for (int r = 0; r < 2; ++r) {
        int cid = tid + r * 128;
        int krow = cid >> 3, kc = cid & 7;
        async_copy16(&sK[cur ^ 1][krow * 64 + kc * 8],
                     Kh + (size_t)(nb + krow) * DH_ + kc * 8);
        int vrow = cid >> 2, vc = cid & 3;
        async_copy16(&sVt[cur ^ 1][vrow * 32 + vc * 8],
                     Vth + (size_t)vrow * S_ + nb + vc * 8);
      }
    }

    // S = Q K^T
    v8f s0 = {}, s1 = {};
#pragma unroll
    for (int ks = 0; ks < 2; ++ks) {
      FragAB b0, b1;
      b0.q[0] = *(const uint4*)(&sK[cur][(c) * 64 + ks * 32 + half * 8]);
      b0.q[1] = *(const uint4*)(&sK[cur][(c) * 64 + ks * 32 + 16 + half * 8]);
      b1.q[0] = *(const uint4*)(&sK[cur][(16 + c) * 64 + ks * 32 + half * 8]);
      b1.q[1] = *(const uint4*)(&sK[cur][(16 + c) * 64 + ks * 32 + 16 + half * 8]);
      s0 = wmma_bf16(aQ[ks].v, b0.v, s0);
      s1 = wmma_bf16(aQ[ks].v, b1.v, s1);
    }
    const float sc = 0.125f;  // 1/sqrt(DH)
#pragma unroll
    for (int i = 0; i < 8; ++i) { s0[i] *= sc; s1[i] *= sc; }

    if (kvb + 31 > q0) {      // diagonal tiles only
#pragma unroll
      for (int i = 0; i < 8; ++i) {
        int qr = q0 + i + 8 * half;
        if (kvb + c      > qr) s0[i] = -3.0e38f;
        if (kvb + 16 + c > qr) s1[i] = -3.0e38f;
      }
    }

    // online softmax: row reductions across the 16-lane half via shfl_xor
    v8f mnew;
#pragma unroll
    for (int i = 0; i < 8; ++i) {
      float t = fmaxf(s0[i], s1[i]);
#pragma unroll
      for (int d = 1; d < 16; d <<= 1) t = fmaxf(t, __shfl_xor(t, d, 32));
      mnew[i] = fmaxf(mrow[i], t);
    }
    v8f alpha, rs;
#pragma unroll
    for (int i = 0; i < 8; ++i) {
      alpha[i] = __expf(mrow[i] - mnew[i]);
      s0[i] = __expf(s0[i] - mnew[i]);
      s1[i] = __expf(s1[i] - mnew[i]);
      float t = s0[i] + s1[i];
#pragma unroll
      for (int d = 1; d < 16; d <<= 1) t += __shfl_xor(t, d, 32);
      rs[i] = t;
    }
#pragma unroll
    for (int i = 0; i < 8; ++i) {
      lrow[i] = lrow[i] * alpha[i] + rs[i];
      mrow[i] = mnew[i];
      O[0][i] *= alpha[i]; O[1][i] *= alpha[i];
      O[2][i] *= alpha[i]; O[3][i] *= alpha[i];
    }

    // C-layout -> A-layout for P via per-wave LDS staging
    unsigned short* pw = sP[w];
#pragma unroll
    for (int i = 0; i < 8; ++i) {
      int row = i + 8 * half;
      pw[row * 32 + c]      = f2bf(s0[i]);
      pw[row * 32 + 16 + c] = f2bf(s1[i]);
    }
    FragAB aP;
    {
      const unsigned short* pr = pw + (size_t)c * 32;
      aP.q[0] = *(const uint4*)(pr + half * 8);
      aP.q[1] = *(const uint4*)(pr + 16 + half * 8);
    }
#pragma unroll
    for (int t = 0; t < 4; ++t) {
      FragAB bV;
      const unsigned short* vr = &sVt[cur][(size_t)(t * 16 + c) * 32];
      bV.q[0] = *(const uint4*)(vr + half * 8);
      bV.q[1] = *(const uint4*)(vr + 16 + half * 8);
      O[t] = wmma_bf16(aP.v, bV.v, O[t]);
    }

    if (pf) wait_async0();           // next KV tile resident
    __syncthreads();
  }

  // normalize and write ctx in [B][S][D] bf16 layout
  const int bb = bh >> 4, h = bh & (H_ - 1);
  v8f inv;
#pragma unroll
  for (int i = 0; i < 8; ++i) inv[i] = 1.0f / lrow[i];
#pragma unroll
  for (int t = 0; t < 4; ++t)
#pragma unroll
    for (int i = 0; i < 8; ++i) {
      int s   = q0 + i + 8 * half;
      int col = h * 64 + t * 16 + c;
      ctx[((size_t)bb * S_ + s) * D_ + col] = f2bf(O[t][i] * inv[i]);
    }
}

// ----------------------------------------------------------------- launch ---
extern "C" void kernel_launch(void* const* d_in, const int* in_sizes, int n_in,
                              void* d_out, int out_size, void* d_ws, size_t ws_size,
                              hipStream_t stream) {
  const float* x  = (const float*)d_in[0];
  const float* Wq = (const float*)d_in[1];
  const float* Wk = (const float*)d_in[2];
  const float* Wv = (const float*)d_in[3];
  const float* Wo = (const float*)d_in[4];
  const float* bo = (const float*)d_in[5];
  float* out = (float*)d_out;
  unsigned short* ws = (unsigned short*)d_ws;

  const size_t NX = (size_t)B_ * S_ * D_;  // 4,194,304
  const size_t NW = (size_t)D_ * D_;       // 1,048,576
  unsigned short* xb  = ws;
  unsigned short* wqT = xb  + NX;
  unsigned short* wkT = wqT + NW;
  unsigned short* wvT = wkT + NW;
  unsigned short* woT = wvT + NW;
  unsigned short* Qb  = woT + NW;
  unsigned short* Kb  = Qb  + NX;
  unsigned short* VtB = Kb  + NX;
  unsigned short* Cb  = VtB + NX;

  k_convert_bf16<<<dim3((unsigned)(NX / 8 / 256)), 256, 0, stream>>>(x, xb, (int)(NX / 8));
  dim3 gt(32, 32);
  k_convT_bf16<<<gt, 256, 0, stream>>>(Wq, wqT);
  k_convT_bf16<<<gt, 256, 0, stream>>>(Wk, wkT);
  k_convT_bf16<<<gt, 256, 0, stream>>>(Wv, wvT);
  k_convT_bf16<<<gt, 256, 0, stream>>>(Wo, woT);

  dim3 gg(32, 8);  // M/128, N/128
  k_gemm<<<gg, 256, 0, stream>>>(xb, wqT, nullptr, Qb,  nullptr, 0);
  k_gemm<<<gg, 256, 0, stream>>>(xb, wkT, nullptr, Kb,  nullptr, 0);
  k_gemm<<<gg, 256, 0, stream>>>(xb, wvT, nullptr, VtB, nullptr, 2);

  k_attn<<<dim3(32, 32), 128, 0, stream>>>(Qb, Kb, VtB, Cb);

  k_gemm<<<gg, 256, 0, stream>>>(Cb, woT, bo, nullptr, out, 1);
}
